// ProposalModule_68461778698780
// MI455X (gfx1250) — matrix operand order repaired and checked by reading, not compiled, not run on hardware
//
#include <hip/hip_runtime.h>
#include <hip/hip_bf16.h>

typedef _Float16 h8v  __attribute__((ext_vector_type(8)));
typedef _Float16 h16v __attribute__((ext_vector_type(16)));
typedef float    f8v  __attribute__((ext_vector_type(8)));

#define BB 32
#define NN 2048
#define SS 256
#define NSAMP 16
#define CIN 256
#define KPAD 288      // 259 padded to multiple of 32
#define RADIUS 0.3f
#define R2 0.09f
#define NPTS 8192     // B*S
#define MROWS 131072  // B*S*NS

// ---------------------------------------------------------------------------
// Parameter prep: fold BN into per-channel scale/shift
// ---------------------------------------------------------------------------
__global__ void prep_params_kernel(
    const float* __restrict__ sg, const float* __restrict__ sb,
    const float* __restrict__ sm, const float* __restrict__ sv,
    const float* __restrict__ c1b, const float* __restrict__ b1g,
    const float* __restrict__ b1b, const float* __restrict__ b1m, const float* __restrict__ b1v,
    const float* __restrict__ c2b, const float* __restrict__ b2g,
    const float* __restrict__ b2b, const float* __restrict__ b2m, const float* __restrict__ b2v,
    float* __restrict__ saS, float* __restrict__ saT,
    float* __restrict__ s1, float* __restrict__ t1,
    float* __restrict__ s2, float* __restrict__ t2)
{
    int c = threadIdx.x;
    if (c >= 128) return;
    for (int l = 0; l < 3; ++l) {
        float s = sg[l*128+c] * rsqrtf(sv[l*128+c] + 1e-5f);
        saS[l*128+c] = s;
        saT[l*128+c] = sb[l*128+c] - sm[l*128+c] * s;
    }
    float a = b1g[c] * rsqrtf(b1v[c] + 1e-5f);
    s1[c] = a; t1[c] = (c1b[c] - b1m[c]) * a + b1b[c];
    float d = b2g[c] * rsqrtf(b2v[c] + 1e-5f);
    s2[c] = d; t2[c] = (c2b[c] - b2m[c]) * d + b2b[c];
}

// f32 [O x Cin] -> f16 [O x Kp] (zero pad)
__global__ void cvt_kernel(const float* __restrict__ src, _Float16* __restrict__ dst,
                           int O, int Cin, int Kp)
{
    int i = blockIdx.x * 256 + threadIdx.x;
    if (i >= O * Kp) return;
    int o = i / Kp, k = i - o * Kp;
    dst[i] = (_Float16)(k < Cin ? src[o * Cin + k] : 0.0f);
}

// ---------------------------------------------------------------------------
// Furthest point sampling: one block per batch, point cloud in LDS
// ---------------------------------------------------------------------------
__global__ __launch_bounds__(256) void fps_kernel(const float* __restrict__ xyz,
                                                  float* __restrict__ new_xyz)
{
    __shared__ float sx[NN], sy[NN], sz[NN], sd[NN];
    __shared__ float rv[256];
    __shared__ int   ri[256];
    __shared__ int   sfar;
    const int b = blockIdx.x, tid = threadIdx.x;
    const float* p = xyz + (size_t)b * NN * 3;
    for (int k = tid; k < NN; k += 256) {
        sx[k] = p[k*3+0]; sy[k] = p[k*3+1]; sz[k] = p[k*3+2];
        sd[k] = 1e10f;
    }
    if (tid == 0) sfar = 0;
    __syncthreads();
    for (int s = 0; s < SS; ++s) {
        int far = sfar;
        if (tid == 0) {
            float* o = new_xyz + ((size_t)b * SS + s) * 3;
            o[0] = sx[far]; o[1] = sy[far]; o[2] = sz[far];
        }
        float cx = sx[far], cy = sy[far], cz = sz[far];
        float best = -1.0f; int bi = 0;
        for (int k = tid; k < NN; k += 256) {
            float dx = sx[k]-cx, dy = sy[k]-cy, dz = sz[k]-cz;
            float d  = dx*dx + dy*dy + dz*dz;
            float nd = fminf(sd[k], d);
            sd[k] = nd;
            if (nd > best) { best = nd; bi = k; }   // first max (lowest idx) per thread
        }
        rv[tid] = best; ri[tid] = bi;
        __syncthreads();
        for (int off = 128; off > 0; off >>= 1) {
            if (tid < off) {
                float v2 = rv[tid+off]; int i2 = ri[tid+off];
                if (v2 > rv[tid] || (v2 == rv[tid] && i2 < ri[tid])) { rv[tid] = v2; ri[tid] = i2; }
            }
            __syncthreads();
        }
        if (tid == 0) sfar = ri[0];
        __syncthreads();
    }
}

// ---------------------------------------------------------------------------
// Ball query + gather + f16 convert: one wave32 per centroid
// ---------------------------------------------------------------------------
__global__ __launch_bounds__(256) void ballgather_kernel(
    const float* __restrict__ xyz, const float* __restrict__ feat,
    const float* __restrict__ new_xyz, _Float16* __restrict__ Hin)
{
    __shared__ int idxs[8][16];
    const int wave = threadIdx.x >> 5, lane = threadIdx.x & 31;
    const int cidx = blockIdx.x * 8 + wave;          // 0..8191
    const int b = cidx >> 8;
    const float* cen = new_xyz + (size_t)cidx * 3;
    const float cx = cen[0], cy = cen[1], cz = cen[2];
    const float* p = xyz + (size_t)b * NN * 3;

    int count = 0;
    for (int base = 0; base < NN && count < NSAMP; base += 32) {
        int k = base + lane;
        float dx = p[k*3+0]-cx, dy = p[k*3+1]-cy, dz = p[k*3+2]-cz;
        bool in = (dx*dx + dy*dy + dz*dz) < R2;
        unsigned m = __builtin_amdgcn_ballot_w32(in);
        int pre = __builtin_popcount(m & ((1u << lane) - 1u));
        if (in && (count + pre) < NSAMP) idxs[wave][count + pre] = k;
        count += __builtin_popcount(m);
    }
    asm volatile("s_wait_dscnt 0" ::: "memory");
    if (lane < NSAMP) {
        int v;
        if (count == 0)      v = 0;
        else if (lane < count) v = idxs[wave][lane];
        else                 v = idxs[wave][0];        // pad with first hit
        idxs[wave][lane] = v;
    }
    asm volatile("s_wait_dscnt 0" ::: "memory");

    const float* fb = feat + (size_t)b * CIN * NN;
    for (int n = 0; n < NSAMP; ++n) {
        int id = idxs[wave][n];
        _Float16* row = Hin + ((size_t)cidx * NSAMP + n) * KPAD;
        #pragma unroll
        for (int cc = 0; cc < KPAD / 32; ++cc) {
            int c = cc * 32 + lane;
            float v;
            if (c < 3)        v = (p[id*3+c] - cen[c]) * (1.0f / RADIUS);
            else if (c < 259) v = fb[(size_t)(c-3) * NN + id];
            else              v = 0.0f;
            row[c] = (_Float16)v;
        }
    }
}

// ---------------------------------------------------------------------------
// WMMA GEMM: out[M x NT*16] = A[M x K] * W[NT*16 x K]^T, optional BN+ReLU.
// Weights are staged into LDS once per workgroup with async global->LDS
// copies (ASYNCcnt path); all 8 waves then feed B-frags from LDS.
// ---------------------------------------------------------------------------
template<int K, int NT, bool RELU, bool OUTF16>
__global__ __launch_bounds__(256) void gemm_wmma(
    const _Float16* __restrict__ A, const _Float16* __restrict__ W,
    const float* __restrict__ scale, const float* __restrict__ shift,
    void* __restrict__ outv, int M)
{
    __shared__ __align__(32) _Float16 ldsW[NT * 16 * K];

    // --- cooperative async stage of the whole weight block into LDS ---
    constexpr int NV = (NT * 16 * K) / 8;      // number of 16B transfers
    {
        unsigned lbase =
            (unsigned)(uintptr_t)(__attribute__((address_space(3))) _Float16*)ldsW;
        for (int i = threadIdx.x; i < NV; i += 256) {
            unsigned loff = lbase + (unsigned)i * 16u;
            unsigned long long gaddr = (unsigned long long)(uintptr_t)(W + (size_t)i * 8);
            asm volatile("global_load_async_to_lds_b128 %0, %1, off"
                         :: "v"(loff), "v"(gaddr) : "memory");
        }
        asm volatile("s_wait_asynccnt 0" ::: "memory");
    }
    __syncthreads();

    const int lane = threadIdx.x & 31;
    const int wave = threadIdx.x >> 5;
    const int m0 = (blockIdx.x * 8 + wave) * 16;
    if (m0 >= M) return;

    f8v acc[NT];
    #pragma unroll
    for (int t = 0; t < NT; ++t) acc[t] = f8v{0,0,0,0,0,0,0,0};

    const int am = m0 + (lane & 15);
    const int ak = (lane >> 4) * 16;      // A: lanes 0-15 -> K 0..15; 16-31 -> K 16..31
    const int bn = lane & 15;
    const int bk = (lane >> 4) * 16;      // B: lanes 0-15 -> K 0..15; 16-31 -> K 16..31
    const _Float16* ap = A + (size_t)am * K + ak;
    const _Float16* wl = ldsW + (size_t)bn * K + bk;

    #pragma unroll
    for (int kc = 0; kc < K; kc += 32) {
        h16v a = *(const h16v*)(ap + kc);
        #pragma unroll
        for (int t = 0; t < NT; ++t) {
            h16v bfrag = *(const h16v*)(wl + (size_t)t * 16 * K + kc);
            acc[t] = __builtin_amdgcn_wmma_f32_16x16x32_f16(
                false, a, false, bfrag, (short)0, acc[t], false, false);
        }
    }

    const int mbase = m0 + (lane >> 4) * 8;
    #pragma unroll
    for (int t = 0; t < NT; ++t) {
        #pragma unroll
        for (int j = 0; j < 8; ++j) {
            int m = mbase + j;
            int n = t * 16 + (lane & 15);
            float v = acc[t][j];
            if (scale) v = v * scale[n] + shift[n];
            if (RELU)  v = v > 0.0f ? v : 0.0f;
            if (OUTF16) ((_Float16*)outv)[(size_t)m * (NT*16) + n] = (_Float16)v;
            else        ((float*)outv)[(size_t)m * (NT*16) + n] = v;
        }
    }
}

// ---------------------------------------------------------------------------
// Max over 16 samples per centroid -> x (B,128,S) f32 and xT (pts,128) f16
// ---------------------------------------------------------------------------
__global__ void maxreduce_kernel(const _Float16* __restrict__ H3,
                                 float* __restrict__ x, _Float16* __restrict__ xT)
{
    int i = blockIdx.x * 256 + threadIdx.x;   // NPTS*128
    if (i >= NPTS * 128) return;
    int pIdx = i >> 7, c = i & 127;
    const _Float16* base = H3 + (size_t)pIdx * NSAMP * 128 + c;
    float m = (float)base[0];
    #pragma unroll
    for (int n = 1; n < NSAMP; ++n) m = fmaxf(m, (float)base[n * 128]);
    int b = pIdx >> 8, s = pIdx & 255;
    x[((size_t)b * 128 + c) * SS + s] = m;
    xT[(size_t)pIdx * 128 + c] = (_Float16)m;
}

// ---------------------------------------------------------------------------
// Attention: att[b][g] = sum_{c,s} p*g   (tpg layout: [pt][192] = t|p|g)
// ---------------------------------------------------------------------------
__global__ __launch_bounds__(256) void att_kernel(const float* __restrict__ tpg,
                                                  float* __restrict__ att)
{
    int b = blockIdx.x >> 2, g = blockIdx.x & 3;
    __shared__ float red[256];
    float sum = 0.0f;
    for (int e = threadIdx.x; e < 16 * SS; e += 256) {
        int c = e >> 8, s = e & 255;
        size_t ro = ((size_t)b * SS + s) * 192;
        sum += tpg[ro + 64 + g*16 + c] * tpg[ro + 128 + g*16 + c];
    }
    red[threadIdx.x] = sum;
    __syncthreads();
    for (int off = 128; off > 0; off >>= 1) {
        if (threadIdx.x < off) red[threadIdx.x] += red[threadIdx.x + off];
        __syncthreads();
    }
    if (threadIdx.x == 0) att[blockIdx.x] = red[0];
}

// z[b][g*32+o][s] = att[b][g] * sum_c wz[(g*32+o)*16+c] * t[b][g*16+c][s]
__global__ void z_kernel(const float* __restrict__ tpg, const float* __restrict__ att,
                         const float* __restrict__ wz, float* __restrict__ z)
{
    int i = blockIdx.x * 256 + threadIdx.x;   // B*128*S
    if (i >= BB * 128 * SS) return;
    int s = i & 255, ch = (i >> 8) & 127, b = i >> 15;
    int g = ch >> 5;
    float a = att[b * 4 + g];
    size_t ro = ((size_t)b * SS + s) * 192;
    const float* wr = wz + ch * 16;
    float sum = 0.0f;
    #pragma unroll
    for (int c = 0; c < 16; ++c) sum += wr[c] * tpg[ro + g*16 + c];
    z[i] = a * sum;
}

// GroupNorm over (b,g) slabs of 32x256, + residual x -> yT f16 (pt-major)
__global__ __launch_bounds__(256) void gnres_kernel(
    const float* __restrict__ z, const float* __restrict__ x,
    const float* __restrict__ gamma, const float* __restrict__ beta,
    _Float16* __restrict__ yT)
{
    int b = blockIdx.x >> 2, g = blockIdx.x & 3;
    __shared__ float r1[256], r2[256];
    const float* zb = z + ((size_t)b * 128 + g * 32) * SS;    // 8192 elems
    float s1 = 0.0f, s2 = 0.0f;
    for (int e = threadIdx.x; e < 8192; e += 256) { float v = zb[e]; s1 += v; s2 += v*v; }
    r1[threadIdx.x] = s1; r2[threadIdx.x] = s2;
    __syncthreads();
    for (int off = 128; off > 0; off >>= 1) {
        if (threadIdx.x < off) { r1[threadIdx.x] += r1[threadIdx.x+off]; r2[threadIdx.x] += r2[threadIdx.x+off]; }
        __syncthreads();
    }
    float mu  = r1[0] * (1.0f / 8192.0f);
    float var = r2[0] * (1.0f / 8192.0f) - mu * mu;
    float inv = rsqrtf(var + 1e-5f);
    for (int e = threadIdx.x; e < 8192; e += 256) {
        int o = e >> 8, s = e & 255;
        int ch = g * 32 + o;
        float v = (zb[e] - mu) * inv * gamma[ch] + beta[ch]
                + x[((size_t)b * 128 + ch) * SS + s];
        yT[((size_t)b * SS + s) * 128 + ch] = (_Float16)v;
    }
}

// Final 128->43 conv (tiny) -> netT [pt][43] f32
__global__ void conv3_kernel(const _Float16* __restrict__ h2T, const float* __restrict__ w,
                             const float* __restrict__ bias, float* __restrict__ net)
{
    int i = blockIdx.x * 256 + threadIdx.x;   // NPTS*43
    if (i >= NPTS * 43) return;
    int o = i % 43, pIdx = i / 43;
    const _Float16* a = h2T + (size_t)pIdx * 128;
    const float* wr = w + o * 128;
    float sum = bias[o];
    #pragma unroll
    for (int c = 0; c < 128; ++c) sum += wr[c] * (float)a[c];
    net[(size_t)pIdx * 43 + o] = sum;
}

// Output assembly: (B,S,55)
__global__ void assemble_kernel(const float* __restrict__ net, const float* __restrict__ nz,
                                float* __restrict__ out)
{
    int pIdx = blockIdx.x * 256 + threadIdx.x;
    if (pIdx >= NPTS) return;
    const float* nt = net + (size_t)pIdx * 43;
    const float* c  = nz  + (size_t)pIdx * 3;
    float* o = out + (size_t)pIdx * 55;
    o[0] = nt[0]; o[1] = nt[1];
    o[2] = c[0] + nt[2]; o[3] = c[1] + nt[3]; o[4] = c[2] + nt[4];
    #pragma unroll
    for (int j = 0; j < 26; ++j) o[5 + j] = nt[5 + j];            // nt[5:31]
    const float sc = 3.14159265358979323846f / 12.0f;
    #pragma unroll
    for (int j = 0; j < 12; ++j) o[31 + j] = nt[19 + j] * sc;      // ang_res * pi/NAB
    #pragma unroll
    for (int j = 0; j < 12; ++j) o[43 + j] = nt[31 + j];           // nt[31:43]
}

// ---------------------------------------------------------------------------
extern "C" void kernel_launch(void* const* d_in, const int* in_sizes, int n_in,
                              void* d_out, int out_size, void* d_ws, size_t ws_size,
                              hipStream_t stream)
{
    const float* xyz   = (const float*)d_in[0];
    const float* feat  = (const float*)d_in[1];
    const float* sa_w0 = (const float*)d_in[2];
    const float* sa_w1 = (const float*)d_in[3];
    const float* sa_w2 = (const float*)d_in[4];
    const float* sa_g  = (const float*)d_in[5];
    const float* sa_b  = (const float*)d_in[6];
    const float* sa_m  = (const float*)d_in[7];
    const float* sa_v  = (const float*)d_in[8];
    const float* wt    = (const float*)d_in[9];
    const float* wp    = (const float*)d_in[10];
    const float* wg    = (const float*)d_in[11];
    const float* wz    = (const float*)d_in[12];
    const float* gng   = (const float*)d_in[13];
    const float* gnb   = (const float*)d_in[14];
    const float* c1w   = (const float*)d_in[15];
    const float* c1b   = (const float*)d_in[16];
    const float* b1g   = (const float*)d_in[17];
    const float* b1b   = (const float*)d_in[18];
    const float* b1m   = (const float*)d_in[19];
    const float* b1v   = (const float*)d_in[20];
    const float* c2w   = (const float*)d_in[21];
    const float* c2b   = (const float*)d_in[22];
    const float* b2g   = (const float*)d_in[23];
    const float* b2b   = (const float*)d_in[24];
    const float* b2m   = (const float*)d_in[25];
    const float* b2v   = (const float*)d_in[26];
    const float* c3w   = (const float*)d_in[27];
    const float* c3b   = (const float*)d_in[28];

    // deterministic scratch carve (256B aligned)
    char* ws = (char*)d_ws;
    size_t off = 0;
    auto carve = [&](size_t bytes) -> void* {
        void* p = ws + off;
        off = (off + bytes + 255) & ~(size_t)255;
        return p;
    };
    _Float16* Hin  = (_Float16*)carve((size_t)MROWS * KPAD * 2);
    _Float16* H1   = (_Float16*)carve((size_t)MROWS * 128 * 2);
    _Float16* H2   = (_Float16*)carve((size_t)MROWS * 128 * 2);
    _Float16* xT   = (_Float16*)carve((size_t)NPTS * 128 * 2);
    float*    x    = (float*)   carve((size_t)BB * 128 * SS * 4);
    float*    tpg  = (float*)   carve((size_t)NPTS * 192 * 4);
    float*    att  = (float*)   carve((size_t)BB * 4 * 4);
    float*    z    = (float*)   carve((size_t)BB * 128 * SS * 4);
    _Float16* yT   = (_Float16*)carve((size_t)NPTS * 128 * 2);
    _Float16* h1T  = (_Float16*)carve((size_t)NPTS * 128 * 2);
    _Float16* h2T  = (_Float16*)carve((size_t)NPTS * 128 * 2);
    float*    net  = (float*)   carve((size_t)NPTS * 43 * 4);
    float*    nz   = (float*)   carve((size_t)NPTS * 3 * 4);
    _Float16* w0h  = (_Float16*)carve((size_t)128 * KPAD * 2);
    _Float16* w1h  = (_Float16*)carve((size_t)128 * 128 * 2);
    _Float16* w2h  = (_Float16*)carve((size_t)128 * 128 * 2);
    _Float16* wc1h = (_Float16*)carve((size_t)128 * 128 * 2);
    _Float16* wc2h = (_Float16*)carve((size_t)128 * 128 * 2);
    _Float16* wtpg = (_Float16*)carve((size_t)192 * 128 * 2);
    float*    saS  = (float*)   carve(3 * 128 * 4);
    float*    saT  = (float*)   carve(3 * 128 * 4);
    float*    s1   = (float*)   carve(128 * 4);
    float*    t1   = (float*)   carve(128 * 4);
    float*    s2   = (float*)   carve(128 * 4);
    float*    t2   = (float*)   carve(128 * 4);

    // 1) fold BN params
    prep_params_kernel<<<1, 128, 0, stream>>>(sa_g, sa_b, sa_m, sa_v,
        c1b, b1g, b1b, b1m, b1v, c2b, b2g, b2b, b2m, b2v,
        saS, saT, s1, t1, s2, t2);

    // 2) convert weights to f16 (zero-padded where needed)
    cvt_kernel<<<(128*KPAD + 255)/256, 256, 0, stream>>>(sa_w0, w0h, 128, 259, KPAD);
    cvt_kernel<<<(128*128 + 255)/256, 256, 0, stream>>>(sa_w1, w1h, 128, 128, 128);
    cvt_kernel<<<(128*128 + 255)/256, 256, 0, stream>>>(sa_w2, w2h, 128, 128, 128);
    cvt_kernel<<<(128*128 + 255)/256, 256, 0, stream>>>(c1w, wc1h, 128, 128, 128);
    cvt_kernel<<<(128*128 + 255)/256, 256, 0, stream>>>(c2w, wc2h, 128, 128, 128);
    cvt_kernel<<<(64*128 + 255)/256, 256, 0, stream>>>(wt, wtpg,            64, 128, 128);
    cvt_kernel<<<(64*128 + 255)/256, 256, 0, stream>>>(wp, wtpg + 64*128,   64, 128, 128);
    cvt_kernel<<<(64*128 + 255)/256, 256, 0, stream>>>(wg, wtpg + 128*128,  64, 128, 128);

    // 3) FPS -> new_xyz
    fps_kernel<<<BB, 256, 0, stream>>>(xyz, nz);

    // 4) ball query + gather -> Hin (f16, K padded to 288)
    ballgather_kernel<<<NPTS / 8, 256, 0, stream>>>(xyz, feat, nz, Hin);

    // 5) SA MLP: three WMMA GEMMs with fused BN+ReLU, f16 out
    gemm_wmma<KPAD, 8, true, true><<<MROWS/(16*8), 256, 0, stream>>>(Hin, w0h, saS,       saT,       (void*)H1, MROWS);
    gemm_wmma<128,  8, true, true><<<MROWS/(16*8), 256, 0, stream>>>(H1,  w1h, saS + 128, saT + 128, (void*)H2, MROWS);
    gemm_wmma<128,  8, true, true><<<MROWS/(16*8), 256, 0, stream>>>(H2,  w2h, saS + 256, saT + 256, (void*)H1, MROWS); // H3 in H1

    // 6) max over 16 samples -> x (B,128,S) f32 + xT f16
    maxreduce_kernel<<<(NPTS*128 + 255)/256, 256, 0, stream>>>(H1, x, xT);

    // 7) t|p|g projection: WMMA GEMM, N=192, f32 out, no epilogue
    gemm_wmma<128, 12, false, false><<<NPTS/(16*8), 256, 0, stream>>>(xT, wtpg, nullptr, nullptr, (void*)tpg, NPTS);

    // 8) grouped global attention
    att_kernel<<<BB * 4, 256, 0, stream>>>(tpg, att);
    z_kernel<<<(BB*128*SS + 255)/256, 256, 0, stream>>>(tpg, att, wz, z);

    // 9) GroupNorm + residual -> yT f16
    gnres_kernel<<<BB * 4, 256, 0, stream>>>(z, x, gng, gnb, yT);

    // 10) conv1/conv2: WMMA GEMMs with fused BN+ReLU
    gemm_wmma<128, 8, true, true><<<NPTS/(16*8), 256, 0, stream>>>(yT,  wc1h, s1, t1, (void*)h1T, NPTS);
    gemm_wmma<128, 8, true, true><<<NPTS/(16*8), 256, 0, stream>>>(h1T, wc2h, s2, t2, (void*)h2T, NPTS);

    // 11) final 128->43 conv + bias
    conv3_kernel<<<(NPTS*43 + 255)/256, 256, 0, stream>>>(h2T, c3w, c3b, net);

    // 12) output assembly (B,S,55)
    assemble_kernel<<<(NPTS + 255)/256, 256, 0, stream>>>(net, nz, (float*)d_out);
}
